// BCAM_49495203119370
// MI455X (gfx1250) — matrix operand order repaired and checked by reading, not compiled, not run on hardware
//
#include <hip/hip_runtime.h>
#include <hip/hip_bf16.h>
#include <math.h>

// CDNA5 wave32 WMMA types
typedef __attribute__((ext_vector_type(16))) __bf16 v16bf;
typedef __attribute__((ext_vector_type(8)))  float  v8f;

#define BB    8
#define CC    192
#define HH    128
#define WW    128
#define HWV   16384     // H*W
#define HEADS 8
#define CHD   24        // channels per head
#define DD    3072      // token dim = CHD * 128
#define NTOK  128

static __device__ __forceinline__ __bf16 tobf(float x) { return (__bf16)x; }

// Branch-free exact-ish GELU: erf via Abramowitz-Stegun 7.1.26 + __expf.
static __device__ __forceinline__ float gelu_erf(float x)
{
    float xs = x * 0.70710678118f;
    float ax = fabsf(xs);
    float t  = 1.0f / (1.0f + 0.3275911f * ax);
    float p  = t * (0.254829592f + t * (-0.284496736f + t * (1.421413741f +
               t * (-1.453152027f + t * 1.061405429f))));
    float e  = 1.0f - p * __expf(-ax * ax);
    e = copysignf(e, xs);
    return 0.5f * x * (1.0f + e);
}

// ---------------------------------------------------------------------------
// conv1x1 as GEMM:  Y[b,o,s] = sum_k Wm[o,k] * X[b,k,s] + bias[o]
// One wave per 16(o) x 64(s) macro-tile: 4 accumulators share each A-frag.
// MODE 0 = plain, 1 = GELU epilogue, 2 = + x1 + x2 residual epilogue.
// ---------------------------------------------------------------------------
template <int MODE>
__global__ __launch_bounds__(256)
void gemm_conv(const float* __restrict__ X, const float* __restrict__ Wm,
               const float* __restrict__ bias, float* __restrict__ Y,
               const float* __restrict__ r1, const float* __restrict__ r2)
{
    const int lane = threadIdx.x & 31;
    const int wv   = threadIdx.x >> 5;
    const int row  = lane & 15;
    const int hi   = lane >> 4;

    int gid = blockIdx.x * 8 + wv;                 // global wave id
    int st  = gid & 255;                           // s macro-tile (16384/64)
    int ot  = (gid >> 8) % 12;                     // o-tile (192/16)
    int b   = gid / (256 * 12);
    const int s0 = st * 64, o0 = ot * 16;

    const float* Xb = X + (size_t)b * CC * HWV;

    v8f acc[4] = {};
    for (int kb = 0; kb < CC; kb += 32) {
        // A: W[o0+row, kb + 16 contiguous] — loaded once, reused by 4 WMMAs
        v16bf a;
        const float* ap = Wm + (o0 + row) * CC + kb + hi * 16;
        #pragma unroll
        for (int t = 0; t < 8; ++t) { a[t] = tobf(ap[t]); a[8 + t] = tobf(ap[8 + t]); }

        const float* bbase = Xb + (size_t)(kb + hi * 16) * HWV + s0 + row;
        if (kb + 32 < CC)
            __builtin_prefetch(bbase + (size_t)32 * HWV, 0, 3);   // next K slab
        #pragma unroll
        for (int u = 0; u < 4; ++u) {
            v16bf bf;
            const float* bp = bbase + 16 * u;
            #pragma unroll
            for (int t = 0; t < 16; ++t) bf[t] = tobf(bp[(size_t)t * HWV]);
            acc[u] = __builtin_amdgcn_wmma_f32_16x16x32_bf16(false, a, false, bf,
                                                             (short)0, acc[u], false, false);
        }
    }
    #pragma unroll
    for (int u = 0; u < 4; ++u) {
        #pragma unroll
        for (int r = 0; r < 8; ++r) {
            int o = o0 + r + hi * 8;
            int s = s0 + 16 * u + row;
            size_t idx = ((size_t)b * CC + o) * HWV + s;
            float v = acc[u][r] + bias[o];
            if (MODE == 1)      v = gelu_erf(v);
            else if (MODE == 2) v += r1[idx] + r2[idx];
            Y[idx] = v;
        }
    }
}

// ---------------------------------------------------------------------------
// Row l2 norms of the token matrices (4 sets):
//   set0: h-rows of out1 (k1)   set1: h-rows of out2 (q1)
//   set2: w-rows of out1 (q2)   set3: w-rows of out2 (k2)
// One wave per row (3072 elements), stores 1/max(||r||,eps).
// ---------------------------------------------------------------------------
__global__ __launch_bounds__(256)
void row_norms(const float* __restrict__ o1, const float* __restrict__ o2,
               float* __restrict__ invn)
{
    const int lane = threadIdx.x & 31;
    int gid  = blockIdx.x * 8 + (threadIdx.x >> 5);
    int row  = gid & 127;
    int head = (gid >> 7) & 7;
    int b    = (gid >> 10) & 7;
    int set  = gid >> 13;

    const float* src  = (set == 0 || set == 2) ? o1 : o2;
    const float* base = src + ((size_t)b * CC + head * CHD) * HWV;
    const bool hrow   = set < 2;

    float s = 0.f;
    for (int kk = lane; kk < DD; kk += 32) {
        int ch = kk >> 7, r2 = kk & 127;
        float v = hrow ? base[(size_t)ch * HWV + row * WW + r2]
                       : base[(size_t)ch * HWV + r2 * WW + row];
        s += v * v;
    }
    #pragma unroll
    for (int off = 16; off; off >>= 1) s += __shfl_xor(s, off, 32);
    if (lane == 0) invn[gid] = 1.0f / fmaxf(sqrtf(s), 1e-12f);
}

// ---------------------------------------------------------------------------
// One block per (b, head). DIR=0: h-tokens (attn1), DIR=1: w-tokens (attn2).
// Phase 1: S = Q Kᵀ (WMMA bf16, K=3072) into 64KB LDS (Q-frag reused x4).
// Phase 2: scale by invq*invk/temp + pos_bias, softmax rows in LDS.
// Phase 3: O = P V + Q̂ (P-frag from LDS reused x4), gated write to fusion.
// ---------------------------------------------------------------------------
template <int DIR>
__global__ __launch_bounds__(256)
void attn_kernel(const float* __restrict__ o1, const float* __restrict__ o2,
                 const float* __restrict__ invn, const float* __restrict__ pbias,
                 const float* __restrict__ gatep, float* __restrict__ fusion)
{
    __shared__ float S[NTOK * NTOK];              // 64 KB
    const int lane = threadIdx.x & 31;
    const int wv   = threadIdx.x >> 5;
    const int row  = lane & 15;
    const int hi   = lane >> 4;
    const int head = blockIdx.x & 7;
    const int b    = blockIdx.x >> 3;

    const float* Q  = (DIR == 0) ? o2 : o1;       // q1=tokens(out2), q2=tokens(out1)
    const float* KV = (DIR == 0) ? o1 : o2;       // k1=v1=out1,      k2=v2=out2
    const float* qb  = Q  + ((size_t)b * CC + head * CHD) * HWV;
    const float* kvb = KV + ((size_t)b * CC + head * CHD) * HWV;
    const float* invq = invn + (DIR == 0 ? 1 : 2) * 8192 + (b * 8 + head) * 128;
    const float* invk = invn + (DIR == 0 ? 0 : 3) * 8192 + (b * 8 + head) * 128;

    // ---- Phase 1: raw S = Q K^T (16 macro-tiles of 16x64) -----------------
    for (int mt = wv; mt < 16; mt += 8) {
        int i0 = (mt & 7) * 16;
        int j0 = (mt >> 3) * 64;
        v8f acc[4] = {};
        for (int kblk = 0; kblk < DD; kblk += 32) {
            int ch = kblk >> 7;                    // 32 | 128: block within one channel
            int wb = kblk & 127;
            v16bf a;
            if (DIR == 0) {                        // h-tokens: K contiguous in w
                const float* ap = qb + (size_t)ch * HWV + (i0 + row) * WW + wb + hi * 16;
                #pragma unroll
                for (int t = 0; t < 8; ++t) { a[t] = tobf(ap[t]); a[8 + t] = tobf(ap[8 + t]); }
            } else {                               // w-tokens: K strided by W
                const float* ap = qb + (size_t)ch * HWV + (wb + hi * 16) * WW + (i0 + row);
                #pragma unroll
                for (int t = 0; t < 8; ++t) { a[t] = tobf(ap[t * WW]); a[8 + t] = tobf(ap[(8 + t) * WW]); }
            }
            #pragma unroll
            for (int u = 0; u < 4; ++u) {
                v16bf bf;
                int j = j0 + 16 * u + row;
                if (DIR == 0) {
                    const float* bp = kvb + (size_t)ch * HWV + j * WW + wb + hi * 16;
                    #pragma unroll
                    for (int t = 0; t < 16; ++t) bf[t] = tobf(bp[t]);
                } else {
                    const float* bp = kvb + (size_t)ch * HWV + (wb + hi * 16) * WW + j;
                    #pragma unroll
                    for (int t = 0; t < 16; ++t) bf[t] = tobf(bp[t * WW]);
                }
                acc[u] = __builtin_amdgcn_wmma_f32_16x16x32_bf16(false, a, false, bf,
                                                                 (short)0, acc[u], false, false);
            }
        }
        #pragma unroll
        for (int u = 0; u < 4; ++u)
            #pragma unroll
            for (int r = 0; r < 8; ++r)
                S[(i0 + r + hi * 8) * NTOK + j0 + 16 * u + row] = acc[u][r];
    }
    __syncthreads();

    // ---- Phase 2: scaled softmax ------------------------------------------
    float g  = 1.0f / (1.0f + __expf(-gatep[0]));
    float gw = (DIR == 0) ? g : (1.0f - g);
    if (threadIdx.x < NTOK) {
        int i = threadIdx.x;
        float iq = invq[i];
        float pb = pbias[head];
        int rot = (i * 2) & 127;                  // bank-spreading rotation
        float mx = -1e30f;
        for (int j = 0; j < NTOK; ++j) {
            int jj = (j + rot) & 127;
            float v = S[i * NTOK + jj] * iq * invk[jj] * 100.0f + pb;  // /temperature
            S[i * NTOK + jj] = v;
            mx = fmaxf(mx, v);
        }
        float sum = 0.f;
        for (int j = 0; j < NTOK; ++j) {
            int jj = (j + rot) & 127;
            float e = __expf(S[i * NTOK + jj] - mx);
            S[i * NTOK + jj] = e;
            sum += e;
        }
        float rs = 1.0f / sum;
        for (int j = 0; j < NTOK; ++j) {
            int jj = (j + rot) & 127;
            S[i * NTOK + jj] *= rs;
        }
    }
    __syncthreads();

    // ---- Phase 3: O = P V + Q_hat (8 x 48 macro-tiles of 16x64) -----------
    for (int mt = wv; mt < 8 * 48; mt += 8) {
        int i0 = (mt / 48) * 16;
        int dg = (mt % 48) * 64;                   // 64-wide d group (single channel)
        v8f acc[4] = {};
        #pragma unroll
        for (int kblk = 0; kblk < NTOK; kblk += 32) {
            v16bf a;                               // P-frag from LDS, reused x4
            const float* ap = &S[(i0 + row) * NTOK + kblk + hi * 16];
            #pragma unroll
            for (int t = 0; t < 8; ++t) { a[t] = tobf(ap[t]); a[8 + t] = tobf(ap[8 + t]); }
            #pragma unroll
            for (int u = 0; u < 4; ++u) {
                int dcol = dg + 16 * u + row;
                int ch = dcol >> 7, dr = dcol & 127;
                v16bf bf;
                if (DIR == 0) {                    // V[j,d]: j strided by W
                    const float* bp = kvb + (size_t)ch * HWV + (kblk + hi * 16) * WW + dr;
                    #pragma unroll
                    for (int t = 0; t < 16; ++t) bf[t] = tobf(bp[t * WW]);
                } else {                           // V[j,d]: j contiguous
                    const float* bp = kvb + (size_t)ch * HWV + dr * WW + kblk + hi * 16;
                    #pragma unroll
                    for (int t = 0; t < 16; ++t) bf[t] = tobf(bp[t]);
                }
                acc[u] = __builtin_amdgcn_wmma_f32_16x16x32_bf16(false, a, false, bf,
                                                                 (short)0, acc[u], false, false);
            }
        }
        #pragma unroll
        for (int u = 0; u < 4; ++u) {
            int dcol = dg + 16 * u + row;
            int ch = dcol >> 7, dr = dcol & 127;
            float* fb = fusion + ((size_t)b * CC + head * CHD + ch) * HWV;
            #pragma unroll
            for (int r = 0; r < 8; ++r) {
                int i = i0 + r + hi * 8;
                float qv = (DIR == 0) ? qb[(size_t)ch * HWV + i * WW + dr]
                                      : qb[(size_t)ch * HWV + dr * WW + i];
                float v = acc[u][r] + qv * invq[i];   // + q_hat residual
                size_t idx = (DIR == 0) ? ((size_t)i * WW + dr) : ((size_t)dr * WW + i);
                if (DIR == 0) fb[idx]  = gw * v;      // g * out3
                else          fb[idx] += gw * v;      // + (1-g) * out4
            }
        }
    }
}

// ---------------------------------------------------------------------------
extern "C" void kernel_launch(void* const* d_in, const int* in_sizes, int n_in,
                              void* d_out, int out_size, void* d_ws, size_t ws_size,
                              hipStream_t stream)
{
    const float* x1     = (const float*)d_in[0];
    const float* x2     = (const float*)d_in[1];
    const float* W_proj = (const float*)d_in[2];
    const float* b_proj = (const float*)d_in[3];
    const float* gate   = (const float*)d_in[4];
    const float* pbh    = (const float*)d_in[5];
    const float* pbw    = (const float*)d_in[6];
    const float* W_m1   = (const float*)d_in[7];
    const float* b_m1   = (const float*)d_in[8];
    const float* W_m2   = (const float*)d_in[9];
    const float* b_m2   = (const float*)d_in[10];

    float* ws = (float*)d_ws;
    const size_t NBC = (size_t)BB * CC * HWV;      // 25,165,824 floats
    float* out1   = ws;
    float* out2   = ws + NBC;
    float* fusion = ws + 2 * NBC;
    float* invn   = ws + 3 * NBC;                  // 4 * 8192 floats

    dim3 blk(256);
    const int gemm_blocks = (BB * 12 * 256) / 8;   // 3072 blocks, 24576 waves

    // projections
    gemm_conv<0><<<gemm_blocks, blk, 0, stream>>>(x1, W_proj, b_proj, out1, nullptr, nullptr);
    gemm_conv<0><<<gemm_blocks, blk, 0, stream>>>(x2, W_proj, b_proj, out2, nullptr, nullptr);
    // token-row norms (folded l2norm)
    row_norms<<<4096, blk, 0, stream>>>(out1, out2, invn);
    // bidirectional attention, gate-fused
    attn_kernel<0><<<BB * HEADS, blk, 0, stream>>>(out1, out2, invn, pbh, gate, fusion);
    attn_kernel<1><<<BB * HEADS, blk, 0, stream>>>(out1, out2, invn, pbw, gate, fusion);
    // tail: proj -> gelu(mlp1) -> mlp2 + x1 + x2
    gemm_conv<0><<<gemm_blocks, blk, 0, stream>>>(fusion, W_proj, b_proj, out1, nullptr, nullptr);
    gemm_conv<1><<<gemm_blocks, blk, 0, stream>>>(out1, W_m1, b_m1, out2, nullptr, nullptr);
    gemm_conv<2><<<gemm_blocks, blk, 0, stream>>>(out2, W_m2, b_m2, (float*)d_out, x1, x2);
}
